// AttentionDecoder_81492709474448
// MI455X (gfx1250) — compile-verified, matrix operand order checked by reading, not run on hardware
//
#include <hip/hip_runtime.h>
#include <math.h>

// ---------------------------------------------------------------- types ----
typedef __bf16 bf16;
typedef bf16  v16bf __attribute__((ext_vector_type(16)));
typedef bf16  v8bf  __attribute__((ext_vector_type(8)));
typedef bf16  v4bf  __attribute__((ext_vector_type(4)));
typedef float v8f   __attribute__((ext_vector_type(8)));
typedef float v4f   __attribute__((ext_vector_type(4)));

#define T_DEC 32
#define BATCH 64
#define T_ENC 128
#define SENT  4096
#define EMB   300
#define DEC   512
#define ATT   512
#define VOC   32000
#define XIN_K 1344   // 300 + 512 + 512 = 1324, padded to multiple of 64

// ------------------------------------------------- WMMA fragment loaders ---
// CDNA5 16-bit A/B 16x32 layout (ISA 7.12.2):
//   lanes 0-15 : elems 0-7 = K 0..7,  elems 8-15 = K 16..23   (row = lane)
//   lanes 16-31: elems 0-7 = K 8..15, elems 8-15 = K 24..31   (row = lane-16)
__device__ __forceinline__ v16bf load_frag_bf16(const bf16* p /* row base + k + 8*hi */) {
  v8bf lo = *(const v8bf*)p;
  v8bf up = *(const v8bf*)(p + 16);
  v16bf f;
#pragma unroll
  for (int i = 0; i < 8; ++i) { f[i] = lo[i]; f[i + 8] = up[i]; }
  return f;
}

__device__ __forceinline__ v16bf load_frag_f32(const float* p /* row base + k + 8*hi */) {
  v4f a = *(const v4f*)p;
  v4f b = *(const v4f*)(p + 4);
  v4f c = *(const v4f*)(p + 16);
  v4f d = *(const v4f*)(p + 20);
  v16bf f;
#pragma unroll
  for (int i = 0; i < 4; ++i) {
    f[i]      = (bf16)a[i];
    f[i + 4]  = (bf16)b[i];
    f[i + 8]  = (bf16)c[i];
    f[i + 12] = (bf16)d[i];
  }
  return f;
}

// --------------------------------------------------------- WMMA GEMM ------
// C[M,N] = act( A[M,K] @ W[N,K]^T + bias )  with A in bf16 or f32->bf16.
// Block = 256 threads = 8 waves (2 M-waves x 4 N-waves), block tile 64x128,
// wave tile 32x32; K loop unrolled x2 -> 8 v_wmma_f32_16x16x32_bf16 per iter.
// Fixed per-lane base pointers + k indexing keep the loads on the GLOBAL
// path (loop-carried pointer phis would defeat address-space inference and
// fall back to FLAT, which ties up DScnt / the LDS issue path).
// Requires M % 64 == 0, N % 128 == 0, K % 64 == 0 (guaranteed by padding).
template <bool AF32>
__global__ void __launch_bounds__(256)
gemm_bt_kernel(const void* __restrict__ Av, const bf16* __restrict__ Wt,
               const float* __restrict__ bias, float* __restrict__ Cf,
               bf16* __restrict__ Cb, int M, int N, int K, int act) {
  (void)M;
  const int lane = threadIdx.x & 31;
  const int wave = threadIdx.x >> 5;
  const int hi   = lane >> 4;
  const int lr   = lane & 15;
  const int m0   = blockIdx.y * 64 + (wave >> 2) * 32;
  const int n0   = blockIdx.x * 128 + (wave & 3) * 32;

  v8f acc[2][2];
#pragma unroll
  for (int i = 0; i < 2; ++i)
#pragma unroll
    for (int j = 0; j < 2; ++j) {
      v8f z;
#pragma unroll
      for (int v = 0; v < 8; ++v) z[v] = 0.f;
      acc[i][j] = z;
    }

  // fixed per-lane row base pointers (k-indexed inside the loop)
  const float* aF0 = nullptr; const float* aF1 = nullptr;
  const bf16*  aB0 = nullptr; const bf16*  aB1 = nullptr;
  if (AF32) {
    aF0 = (const float*)Av + (size_t)(m0 + lr) * K + 8 * hi;
    aF1 = (const float*)Av + (size_t)(m0 + 16 + lr) * K + 8 * hi;
  } else {
    aB0 = (const bf16*)Av + (size_t)(m0 + lr) * K + 8 * hi;
    aB1 = (const bf16*)Av + (size_t)(m0 + 16 + lr) * K + 8 * hi;
  }
  const bf16* bB0 = Wt + (size_t)(n0 + lr) * K + 8 * hi;
  const bf16* bB1 = Wt + (size_t)(n0 + 16 + lr) * K + 8 * hi;

  for (int k = 0; k < K; k += 64) {
#pragma unroll
    for (int kk = 0; kk < 64; kk += 32) {
      v16bf a[2], b[2];
      if (AF32) {
        a[0] = load_frag_f32(aF0 + k + kk);
        a[1] = load_frag_f32(aF1 + k + kk);
      } else {
        a[0] = load_frag_bf16(aB0 + k + kk);
        a[1] = load_frag_bf16(aB1 + k + kk);
      }
      b[0] = load_frag_bf16(bB0 + k + kk);
      b[1] = load_frag_bf16(bB1 + k + kk);
#pragma unroll
      for (int i = 0; i < 2; ++i)
#pragma unroll
        for (int j = 0; j < 2; ++j)
          acc[i][j] = __builtin_amdgcn_wmma_f32_16x16x32_bf16(
              false, a[i], false, b[j], (short)0, acc[i][j], false, false);
    }
    if (k + 128 <= K) {  // global_prefetch_b8 for the next K tile
      if (AF32) __builtin_prefetch(aF0 + k + 64, 0, 0);
      else      __builtin_prefetch(aB0 + k + 64, 0, 0);
      __builtin_prefetch(bB0 + k + 64, 0, 0);
    }
  }

  // C layout: VGPR v -> row m0 + 16*i + 8*hi + v ; col n0 + 16*j + lr
  const bool hasBias = (bias != nullptr);
  const bool doTanh  = (act == 1);
  const bool hasF    = (Cf != nullptr);
  const bool hasB    = (Cb != nullptr);
#pragma unroll
  for (int i = 0; i < 2; ++i)
#pragma unroll
    for (int j = 0; j < 2; ++j) {
      const int   col = n0 + 16 * j + lr;
      const float bv  = hasBias ? bias[col] : 0.f;
#pragma unroll
      for (int v = 0; v < 8; ++v) {
        const int row = m0 + 16 * i + 8 * hi + v;
        float val     = acc[i][j][v] + bv;
        if (doTanh) val = tanhf(val);
        const size_t idx = (size_t)row * N + col;
        if (hasF) Cf[idx] = val;
        if (hasB) Cb[idx] = (bf16)val;
      }
    }
}

// ----------------------------------------------------- elementwise utils ---
__global__ void f32_to_bf16_kernel(const float* __restrict__ s, bf16* __restrict__ d, int n) {
  int i = blockIdx.x * blockDim.x + threadIdx.x;
  if (i < n) d[i] = (bf16)s[i];
}

__global__ void zero_bf16_kernel(bf16* __restrict__ d, int n) {
  int i = blockIdx.x * blockDim.x + threadIdx.x;
  if (i < n) d[i] = (bf16)0.f;
}

// rows x scols f32 -> rows x dcols bf16, zero pad columns [scols, dcols)
__global__ void convert_pad_kernel(const float* __restrict__ s, bf16* __restrict__ d,
                                   int rows, int scols, int dcols) {
  int i = blockIdx.x * blockDim.x + threadIdx.x;
  if (i >= rows * dcols) return;
  int r = i / dcols, c = i - r * dcols;
  d[i] = (c < scols) ? (bf16)s[(size_t)r * scols + c] : (bf16)0.f;
}

// ctx_feat[b, :] = [s1 | s2 | |s1-s2| | s1*s2]   -> bf16 [64,16384]
__global__ void build_ctx_kernel(const float* __restrict__ s1, const float* __restrict__ s2,
                                 bf16* __restrict__ ctx) {
  int i = blockIdx.x * blockDim.x + threadIdx.x;
  if (i >= BATCH * SENT) return;
  int b = i / SENT, c = i - b * SENT;
  float a = s1[i], d = s2[i];
  bf16* row = ctx + (size_t)b * (4 * SENT);
  row[c]            = (bf16)a;
  row[SENT + c]     = (bf16)d;
  row[2 * SENT + c] = (bf16)fabsf(a - d);
  row[3 * SENT + c] = (bf16)(a * d);
}

// copy expl[t] (f32 [64,300]) into xin[:, 0:300]
__global__ void prep_x_kernel(const float* __restrict__ et, bf16* __restrict__ xin) {
  int i = blockIdx.x * blockDim.x + threadIdx.x;
  if (i >= BATCH * EMB) return;
  int b = i / EMB, c = i - b * EMB;
  xin[(size_t)b * XIN_K + c] = (bf16)et[i];
}

// ------------------------------------------------------- attention step ---
// q: bf16 [64,1024] (q1|q2, already tanh'ed). kv: bf16 [T_ENC*64, 1024]
// (keys|vals, already tanh'ed).  Writes att into xin[:, 300 + s*512 ...].
__global__ void __launch_bounds__(128)
attention_kernel(const bf16* __restrict__ q, const bf16* __restrict__ kv1,
                 const bf16* __restrict__ kv2, bf16* __restrict__ xin) {
  const int b = blockIdx.x;
  const int s = blockIdx.y;
  const bf16* kv   = s ? kv2 : kv1;
  const bf16* qrow = q + (size_t)b * 1024 + s * ATT;

  __shared__ float qs[ATT];
  __shared__ float sc[T_ENC];
  __shared__ float red[T_ENC];
  const int tid = threadIdx.x;

  for (int h = tid; h < ATT; h += 128) qs[h] = (float)qrow[h];
  __syncthreads();

  // scores: one enc position per thread (T_ENC == 128 == blockDim)
  {
    const v8bf* kp = (const v8bf*)(kv + ((size_t)tid * BATCH + b) * 1024);
    float acc = 0.f;
#pragma unroll 8
    for (int h8 = 0; h8 < ATT / 8; ++h8) {
      v8bf kk = kp[h8];
#pragma unroll
      for (int e = 0; e < 8; ++e) acc += qs[h8 * 8 + e] * (float)kk[e];
    }
    sc[tid]  = acc;
    red[tid] = acc;
  }
  __syncthreads();
  for (int st = 64; st > 0; st >>= 1) {
    if (tid < st) red[tid] = fmaxf(red[tid], red[tid + st]);
    __syncthreads();
  }
  const float mx = red[0];
  __syncthreads();
  const float e = __expf(sc[tid] - mx);
  sc[tid]  = e;
  red[tid] = e;
  __syncthreads();
  for (int st = 64; st > 0; st >>= 1) {
    if (tid < st) red[tid] += red[tid + st];
    __syncthreads();
  }
  const float inv = 1.0f / red[0];
  __syncthreads();

  // att[h] = sum_t a[t] * vals[t,b,h] ; 4 columns per thread (b64 loads)
  const int h0 = tid * 4;
  float a0 = 0.f, a1 = 0.f, a2 = 0.f, a3 = 0.f;
#pragma unroll 4
  for (int t = 0; t < T_ENC; ++t) {
    const float w = sc[t] * inv;
    v4bf vv = *(const v4bf*)(kv + ((size_t)t * BATCH + b) * 1024 + ATT + h0);
    a0 += w * (float)vv[0];
    a1 += w * (float)vv[1];
    a2 += w * (float)vv[2];
    a3 += w * (float)vv[3];
  }
  v4bf out;
  out[0] = (bf16)a0; out[1] = (bf16)a1; out[2] = (bf16)a2; out[3] = (bf16)a3;
  *(v4bf*)(xin + (size_t)b * XIN_K + EMB + s * ATT + h0) = out;
}

// ------------------------------------------------------------- GRU gates ---
__global__ void gru_gate_kernel(const float* __restrict__ gi, const float* __restrict__ gh,
                                float* __restrict__ h, bf16* __restrict__ h_bf,
                                bf16* __restrict__ out_row) {
  int idx = blockIdx.x * blockDim.x + threadIdx.x;
  if (idx >= BATCH * DEC) return;
  const int b = idx >> 9, j = idx & 511;
  const float* gib = gi + (size_t)b * (3 * DEC);
  const float* ghb = gh + (size_t)b * (3 * DEC);
  const float r = 1.f / (1.f + __expf(-(gib[j] + ghb[j])));
  const float z = 1.f / (1.f + __expf(-(gib[DEC + j] + ghb[DEC + j])));
  const float n = tanhf(gib[2 * DEC + j] + r * ghb[2 * DEC + j]);
  const float hv = (1.f - z) * n + z * h[idx];
  h[idx]       = hv;
  h_bf[idx]    = (bf16)hv;
  out_row[idx] = (bf16)hv;
}

// ----------------------------------------------------------------- host ----
static inline dim3 gblk(int n) { return dim3((n + 255) / 256); }

extern "C" void kernel_launch(void* const* d_in, const int* in_sizes, int n_in,
                              void* d_out, int out_size, void* d_ws, size_t ws_size,
                              hipStream_t stream) {
  (void)in_sizes; (void)n_in; (void)out_size; (void)ws_size;
  const float* expl  = (const float*)d_in[0];
  const float* enc1  = (const float*)d_in[1];
  const float* enc2  = (const float*)d_in[2];
  const float* s1    = (const float*)d_in[3];
  const float* s2    = (const float*)d_in[4];
  const float* W_ctx = (const float*)d_in[5];  const float* b_ctx = (const float*)d_in[6];
  const float* Wq1   = (const float*)d_in[7];  const float* bq1   = (const float*)d_in[8];
  const float* Wk1   = (const float*)d_in[9];  const float* bk1   = (const float*)d_in[10];
  const float* Wv1   = (const float*)d_in[11]; const float* bv1   = (const float*)d_in[12];
  const float* Wq2   = (const float*)d_in[13]; const float* bq2   = (const float*)d_in[14];
  const float* Wk2   = (const float*)d_in[15]; const float* bk2   = (const float*)d_in[16];
  const float* Wv2   = (const float*)d_in[17]; const float* bv2   = (const float*)d_in[18];
  const float* W_in  = (const float*)d_in[19]; const float* b_in  = (const float*)d_in[20];
  const float* W_ih  = (const float*)d_in[21]; const float* b_ih  = (const float*)d_in[22];
  const float* W_hh  = (const float*)d_in[23]; const float* b_hh  = (const float*)d_in[24];
  const float* W_voc = (const float*)d_in[25]; const float* b_voc = (const float*)d_in[26];

  // -------- workspace carve-out (aligned 256B) --------
  char*  base = (char*)d_ws;
  size_t off  = 0;
  auto carve = [&](size_t bytes) -> char* {
    char* p = base + off;
    off = (off + bytes + 255) & ~(size_t)255;
    return p;
  };
  bf16*  WQ_bf   = (bf16*)carve((size_t)1024 * DEC * 2);        // [Wq1;Wq2]  [1024,512]
  bf16*  WKV1_bf = (bf16*)carve((size_t)1024 * SENT * 2);       // [Wk1;Wv1]  [1024,4096]
  bf16*  WKV2_bf = (bf16*)carve((size_t)1024 * SENT * 2);
  bf16*  Wctx_bf = (bf16*)carve((size_t)DEC * 4 * SENT * 2);    // [512,16384]
  bf16*  Win_bf  = (bf16*)carve((size_t)DEC * XIN_K * 2);       // [512,1344] padded
  bf16*  Wih_bf  = (bf16*)carve((size_t)3 * DEC * DEC * 2);
  bf16*  Whh_bf  = (bf16*)carve((size_t)3 * DEC * DEC * 2);
  bf16*  Wvoc_bf = (bf16*)carve((size_t)VOC * DEC * 2);
  float* bq_cat  = (float*)carve(1024 * 4);
  float* bkv1    = (float*)carve(1024 * 4);
  float* bkv2    = (float*)carve(1024 * 4);
  bf16*  ctx_bf  = (bf16*)carve((size_t)BATCH * 4 * SENT * 2);  // [64,16384]
  bf16*  kv1_bf  = (bf16*)carve((size_t)T_ENC * BATCH * 1024 * 2);
  bf16*  kv2_bf  = (bf16*)carve((size_t)T_ENC * BATCH * 1024 * 2);
  float* h_f32   = (float*)carve((size_t)BATCH * DEC * 4);
  bf16*  h_bf    = (bf16*)carve((size_t)BATCH * DEC * 2);
  bf16*  q_bf    = (bf16*)carve((size_t)BATCH * 1024 * 2);
  bf16*  xin_bf  = (bf16*)carve((size_t)BATCH * XIN_K * 2);
  bf16*  xp_bf   = (bf16*)carve((size_t)BATCH * DEC * 2);
  float* gi_f    = (float*)carve((size_t)BATCH * 3 * DEC * 4);
  float* gh_f    = (float*)carve((size_t)BATCH * 3 * DEC * 4);
  bf16*  outs_bf = (bf16*)carve((size_t)T_DEC * BATCH * DEC * 2);

  // -------- prepass: weight conversion f32 -> bf16 --------
  f32_to_bf16_kernel<<<gblk(DEC * DEC), 256, 0, stream>>>(Wq1, WQ_bf, DEC * DEC);
  f32_to_bf16_kernel<<<gblk(DEC * DEC), 256, 0, stream>>>(Wq2, WQ_bf + (size_t)DEC * DEC, DEC * DEC);
  f32_to_bf16_kernel<<<gblk(DEC * SENT), 256, 0, stream>>>(Wk1, WKV1_bf, DEC * SENT);
  f32_to_bf16_kernel<<<gblk(DEC * SENT), 256, 0, stream>>>(Wv1, WKV1_bf + (size_t)DEC * SENT, DEC * SENT);
  f32_to_bf16_kernel<<<gblk(DEC * SENT), 256, 0, stream>>>(Wk2, WKV2_bf, DEC * SENT);
  f32_to_bf16_kernel<<<gblk(DEC * SENT), 256, 0, stream>>>(Wv2, WKV2_bf + (size_t)DEC * SENT, DEC * SENT);
  f32_to_bf16_kernel<<<gblk(DEC * 4 * SENT), 256, 0, stream>>>(W_ctx, Wctx_bf, DEC * 4 * SENT);
  f32_to_bf16_kernel<<<gblk(3 * DEC * DEC), 256, 0, stream>>>(W_ih, Wih_bf, 3 * DEC * DEC);
  f32_to_bf16_kernel<<<gblk(3 * DEC * DEC), 256, 0, stream>>>(W_hh, Whh_bf, 3 * DEC * DEC);
  f32_to_bf16_kernel<<<gblk(VOC * DEC), 256, 0, stream>>>(W_voc, Wvoc_bf, VOC * DEC);
  convert_pad_kernel<<<gblk(DEC * XIN_K), 256, 0, stream>>>(W_in, Win_bf, DEC, 2 * ATT + EMB, XIN_K);
  zero_bf16_kernel<<<gblk(BATCH * XIN_K), 256, 0, stream>>>(xin_bf, BATCH * XIN_K);

  // concatenated biases (stay f32)
  hipMemcpyAsync(bq_cat,       bq1, DEC * 4, hipMemcpyDeviceToDevice, stream);
  hipMemcpyAsync(bq_cat + DEC, bq2, DEC * 4, hipMemcpyDeviceToDevice, stream);
  hipMemcpyAsync(bkv1,         bk1, DEC * 4, hipMemcpyDeviceToDevice, stream);
  hipMemcpyAsync(bkv1 + DEC,   bv1, DEC * 4, hipMemcpyDeviceToDevice, stream);
  hipMemcpyAsync(bkv2,         bk2, DEC * 4, hipMemcpyDeviceToDevice, stream);
  hipMemcpyAsync(bkv2 + DEC,   bv2, DEC * 4, hipMemcpyDeviceToDevice, stream);

  // -------- h0 = ctx_feat @ W_ctx^T + b_ctx --------
  build_ctx_kernel<<<gblk(BATCH * SENT), 256, 0, stream>>>(s1, s2, ctx_bf);
  gemm_bt_kernel<false><<<dim3(DEC / 128, BATCH / 64), 256, 0, stream>>>(
      ctx_bf, Wctx_bf, b_ctx, h_f32, h_bf, BATCH, DEC, 4 * SENT, 0);

  // -------- encoder keys|vals (fused, tanh), f32 A converted in-kernel ----
  gemm_bt_kernel<true><<<dim3(1024 / 128, (T_ENC * BATCH) / 64), 256, 0, stream>>>(
      enc1, WKV1_bf, bkv1, nullptr, kv1_bf, T_ENC * BATCH, 1024, SENT, 1);
  gemm_bt_kernel<true><<<dim3(1024 / 128, (T_ENC * BATCH) / 64), 256, 0, stream>>>(
      enc2, WKV2_bf, bkv2, nullptr, kv2_bf, T_ENC * BATCH, 1024, SENT, 1);

  // -------- recurrent decode: 32 steps --------
  for (int t = 0; t < T_DEC; ++t) {
    // q1|q2 = tanh(h @ [Wq1;Wq2]^T + [bq1;bq2])
    gemm_bt_kernel<false><<<dim3(1024 / 128, 1), 256, 0, stream>>>(
        h_bf, WQ_bf, bq_cat, nullptr, q_bf, BATCH, 1024, DEC, 1);
    // xin[:, 0:300] = expl[t]
    prep_x_kernel<<<gblk(BATCH * EMB), 256, 0, stream>>>(expl + (size_t)t * BATCH * EMB, xin_bf);
    // dual attention -> xin[:, 300:1324]
    attention_kernel<<<dim3(BATCH, 2), 128, 0, stream>>>(q_bf, kv1_bf, kv2_bf, xin_bf);
    // xp = xin @ W_in^T + b_in
    gemm_bt_kernel<false><<<dim3(DEC / 128, 1), 256, 0, stream>>>(
        xin_bf, Win_bf, b_in, nullptr, xp_bf, BATCH, DEC, XIN_K, 0);
    // gi = xp @ W_ih^T + b_ih ; gh = h @ W_hh^T + b_hh
    gemm_bt_kernel<false><<<dim3(3 * DEC / 128, 1), 256, 0, stream>>>(
        xp_bf, Wih_bf, b_ih, gi_f, nullptr, BATCH, 3 * DEC, DEC, 0);
    gemm_bt_kernel<false><<<dim3(3 * DEC / 128, 1), 256, 0, stream>>>(
        h_bf, Whh_bf, b_hh, gh_f, nullptr, BATCH, 3 * DEC, DEC, 0);
    // gates -> h, outs[t]
    gru_gate_kernel<<<gblk(BATCH * DEC), 256, 0, stream>>>(
        gi_f, gh_f, h_f32, h_bf, outs_bf + (size_t)t * BATCH * DEC);
  }

  // -------- logits = outs @ W_voc^T + b_voc --------
  gemm_bt_kernel<false><<<dim3(VOC / 128, (T_DEC * BATCH) / 64), 256, 0, stream>>>(
      outs_bf, Wvoc_bf, b_voc, (float*)d_out, nullptr, T_DEC * BATCH, VOC, DEC, 0);
}
